// EdgeMessage_43602507989841
// MI455X (gfx1250) — matrix operand order repaired and checked by reading, not compiled, not run on hardware
//
#include <hip/hip_runtime.h>
#include <hip/hip_bf16.h>

// ---------------------------------------------------------------------------
// GNN layer on gfx1250 (MI455X): leaky slope == 1.0 -> pure affine chains.
// bf16 WMMA (v_wmma_f32_16x16x32_bf16) with f32 accumulate, f32 atomics for
// the edge->node scatter-sum.
// ---------------------------------------------------------------------------

typedef __attribute__((ext_vector_type(16))) __bf16 v16bf;
typedef __attribute__((ext_vector_type(4)))  __bf16 bf16x4;
typedef __attribute__((ext_vector_type(8)))  float  v8f;
typedef __attribute__((ext_vector_type(4)))  float  f32x4;

union Frag { v16bf v; f32x4 f[2]; };

#define NNODES 50000
#define NEDGES 600000
#define TILE   128

// Stage a bf16 weight matrix W[K][ncols] (row-major, global) into per-lane
// WMMA B-fragment order: block (ct,kt) holds 32 lanes x 16 bf16 contiguous.
// B layout (16x16x32 bf16): lane = hi*16 + n ; elements j=0..7 are
// B[kt*32 + hi*8 + j][ct*16 + n], elements 8..15 are the same + 16.
__device__ inline void stage_bfrag(__bf16* __restrict__ F,
                                   const __bf16* __restrict__ Wg,
                                   int ncols, int nkt, int tid) {
  int nct = ncols >> 4;
  int total = nct * nkt * 512;
  for (int idx = tid; idx < total; idx += 256) {
    int j   = idx & 15;
    int l   = (idx >> 4) & 31;
    int blk = idx >> 9;            // blk = ct*nkt + kt
    int kt  = blk % nkt;
    int ct  = blk / nkt;
    int k = kt * 32 + (l >> 4) * 8 + (j & 7) + (j >> 3) * 16;
    int n = ct * 16 + (l & 15);
    F[idx] = Wg[k * ncols + n];
  }
}

// One 16x16 output tile of A[128 x (nkt*32)] @ B, A row-major bf16 in LDS,
// B pre-staged in fragment order. A layout (16-bit 16x32): lane = hi*16 + m,
// elements 0..7 = A[m][k0 + hi*8 .. +7], elements 8..15 = same + 16.
__device__ inline v8f gemm_tile(const __bf16* __restrict__ A, int lda,
                                const __bf16* __restrict__ F, int nkt,
                                int mt, int ct, int lane) {
  int m  = lane & 15;
  int hi = lane >> 4;
  v8f acc = {0.f, 0.f, 0.f, 0.f, 0.f, 0.f, 0.f, 0.f};
  const __bf16* arow = A + (mt * 16 + m) * lda + hi * 8;
  const __bf16* fb   = F + (ct * nkt) * 512 + lane * 16;
#pragma unroll
  for (int kt = 0; kt < nkt; ++kt) {
    Frag a, b;
    a.f[0] = *(const f32x4*)(arow + kt * 32);
    a.f[1] = *(const f32x4*)(arow + kt * 32 + 16);
    b.f[0] = *(const f32x4*)(fb + kt * 512);
    b.f[1] = *(const f32x4*)(fb + kt * 512 + 8);
    acc = __builtin_amdgcn_wmma_f32_16x16x32_bf16(
        false, a.v, false, b.v, (short)0, acc, false, false);
  }
  return acc;
}

// Convert a 16-B f32 quad to bf16 and store as one 8-B LDS write.
__device__ inline void store_bf4(__bf16* __restrict__ d, f32x4 v) {
  bf16x4 t;
  t.x = (__bf16)v.x; t.y = (__bf16)v.y; t.z = (__bf16)v.z; t.w = (__bf16)v.w;
  *(bf16x4*)d = t;
}

// ---------------------------------------------------------------------------
__global__ void zero_msg_kernel(f32x4* __restrict__ p, int n4) {
  int i = blockIdx.x * blockDim.x + threadIdx.x;
  if (i < n4) p[i] = (f32x4){0.f, 0.f, 0.f, 0.f};
}

__global__ void prep_weights_kernel(
    const float* __restrict__ W_nl1, const float* __restrict__ W_el,
    const float* __restrict__ W1,   const float* __restrict__ W2,
    const float* __restrict__ W_nl2, const float* __restrict__ W_ml,
    const float* __restrict__ W_em,
    const float* __restrict__ b_nl1, const float* __restrict__ b_el,
    const float* __restrict__ b_nl2, const float* __restrict__ b_ml,
    __bf16* __restrict__ Wz, __bf16* __restrict__ W1b, __bf16* __restrict__ W2b,
    __bf16* __restrict__ Whb, __bf16* __restrict__ Wemb,
    float* __restrict__ bz, float* __restrict__ bh) {
  int t = blockIdx.x * blockDim.x + threadIdx.x;
  if (t < 256 * 64) {                     // concatenated [K=256][64]
    int k = t >> 6, n = t & 63;
    Wz[t]  = (__bf16)(k < 128 ? W_nl1[k * 64 + n] : W_el[(k - 128) * 64 + n]);
    Whb[t] = (__bf16)(k < 128 ? W_nl2[k * 64 + n] : W_ml[(k - 128) * 64 + n]);
  }
  if (t < 64 * 64)  W1b[t]  = (__bf16)W1[t];
  if (t < 64 * 128) { W2b[t] = (__bf16)W2[t]; Wemb[t] = (__bf16)W_em[t]; }
  if (t < 64) { bz[t] = b_nl1[t] + b_el[t]; bh[t] = b_nl2[t] + b_ml[t]; }
}

// ---------------------------------------------------------------------------
// Edge kernel: per block, 128 edges. Three fused GEMMs + atomic scatter.
// ---------------------------------------------------------------------------
__global__ __launch_bounds__(256) void edge_kernel(
    const float* __restrict__ x, const float* __restrict__ efeat,
    const int* __restrict__ src, const int* __restrict__ dst,
    const __bf16* __restrict__ Wz, const __bf16* __restrict__ W1g,
    const __bf16* __restrict__ W2g,
    const float* __restrict__ bz, const float* __restrict__ b1,
    const float* __restrict__ b2,
    float* __restrict__ msg, int E_) {
  extern __shared__ char smem[];
  __bf16* A    = (__bf16*)smem;               // 128x256 bf16 (64 KB)
  float*  Enew = (float*)smem;                // aliases A: 128x128 f32 (64 KB)
  __bf16* WzF  = (__bf16*)(smem + 65536);     // 32 KB
  __bf16* W1F  = WzF + 16384;                 // 8 KB
  __bf16* W2F  = W1F + 4096;                  // 16 KB
  __bf16* Z    = W2F + 8192;                  // 128x64 bf16 (16 KB)
  __bf16* T    = Z + 8192;                    // 128x64 bf16 (16 KB)
  int*    sidx = (int*)(T + 8192);            // 128
  int*    didx = sidx + 128;                  // 128
  float*  bzs  = (float*)(didx + 128);        // 64
  float*  b1s  = bzs + 64;                    // 64
  float*  b2s  = b1s + 64;                    // 128   -> total 157,696 B

  int tid  = threadIdx.x;
  int lane = tid & 31;
  // Wave id is uniform: force into an SGPR so tile loops use scalar control
  // and EXEC is statically all-ones around every WMMA.
  int w    = __builtin_amdgcn_readfirstlane(tid >> 5);
  int e0   = blockIdx.x * TILE;

  if (tid < TILE) {
    int gi = e0 + tid;
    sidx[tid] = (gi < E_) ? src[gi] : 0;
    didx[tid] = (gi < E_) ? dst[gi] : 0;
  }
  if (tid < 64)  { bzs[tid] = bz[tid]; b1s[tid] = b1[tid]; }
  if (tid < 128) b2s[tid] = b2[tid];
  stage_bfrag(WzF, Wz, 64, 8, tid);   // [256x64]
  stage_bfrag(W1F, W1g, 64, 2, tid);  // [64x64]
  stage_bfrag(W2F, W2g, 128, 2, tid); // [64x128]
  __syncthreads();

  // Gather A = [ x[src] | e ] as bf16: 16-B global loads, 8-B packed LDS stores.
  for (int idx = tid; idx < TILE * 64; idx += 256) {
    int r = idx >> 6, c4 = idx & 63;
    int gi = e0 + r;
    f32x4 v = (f32x4){0.f, 0.f, 0.f, 0.f};
    if (gi < E_) {
      if (c4 < 32) v = ((const f32x4*)x)[(size_t)sidx[r] * 32 + c4];
      else         v = ((const f32x4*)efeat)[(size_t)gi * 32 + (c4 - 32)];
    }
    store_bf4(&A[r * 256 + c4 * 4], v);
  }
  __syncthreads();

  // GEMM1: Z[128x64] = A[128x256] @ Wz + bz   (leaky == identity)
  for (int tile = w; tile < 32; tile += 8) {
    int mt = tile >> 2, ct = tile & 3;
    v8f acc = gemm_tile(A, 256, WzF, 8, mt, ct, lane);
    int ln = lane & 15, hi = lane >> 4, n = ct * 16 + ln;
    float bv = bzs[n];
#pragma unroll
    for (int r = 0; r < 8; ++r)
      Z[(mt * 16 + r + 8 * hi) * 64 + n] = (__bf16)(acc[r] + bv);
  }
  __syncthreads();

  // GEMM2: T = Z @ W1 + b1
  for (int tile = w; tile < 32; tile += 8) {
    int mt = tile >> 2, ct = tile & 3;
    v8f acc = gemm_tile(Z, 64, W1F, 2, mt, ct, lane);
    int ln = lane & 15, hi = lane >> 4, n = ct * 16 + ln;
    float bv = b1s[n];
#pragma unroll
    for (int r = 0; r < 8; ++r)
      T[(mt * 16 + r + 8 * hi) * 64 + n] = (__bf16)(acc[r] + bv);
  }
  __syncthreads();

  // GEMM3: Enew[128x128] = T @ W2 + b2 (f32, aliases A — A is dead here)
  for (int tile = w; tile < 64; tile += 8) {
    int mt = tile >> 3, ct = tile & 7;
    v8f acc = gemm_tile(T, 64, W2F, 2, mt, ct, lane);
    int ln = lane & 15, hi = lane >> 4, n = ct * 16 + ln;
    float bv = b2s[n];
#pragma unroll
    for (int r = 0; r < 8; ++r)
      Enew[(mt * 16 + r + 8 * hi) * 128 + n] = acc[r] + bv;
  }
  __syncthreads();

  // Scatter-sum into msg[dst] with f32 atomics (L2-resident target).
  for (int idx = tid; idx < TILE * 128; idx += 256) {
    int r = idx >> 7, c = idx & 127;
    if (e0 + r < E_)
      atomicAdd(&msg[(size_t)didx[r] * 128 + c], Enew[idx]);
  }
}

// ---------------------------------------------------------------------------
// Node kernel: out = ([x | msg] @ Wh + bh) @ W_em + b_em
// ---------------------------------------------------------------------------
__global__ __launch_bounds__(256) void node_kernel(
    const float* __restrict__ x, const float* __restrict__ msg,
    const __bf16* __restrict__ Whg, const __bf16* __restrict__ Wemg,
    const float* __restrict__ bh, const float* __restrict__ bem,
    float* __restrict__ out, int N_) {
  extern __shared__ char smem[];
  __bf16* A    = (__bf16*)smem;               // 128x256 bf16 (64 KB)
  __bf16* WhF  = (__bf16*)(smem + 65536);     // 32 KB
  __bf16* WemF = WhF + 16384;                 // 16 KB
  __bf16* H    = WemF + 8192;                 // 128x64 bf16 (16 KB)
  float*  bhs  = (float*)(H + 8192);          // 64
  float*  bes  = bhs + 64;                    // 128  -> total 131,840 B

  int tid  = threadIdx.x;
  int lane = tid & 31;
  int w    = __builtin_amdgcn_readfirstlane(tid >> 5);
  int n0   = blockIdx.x * TILE;

  if (tid < 64)  bhs[tid] = bh[tid];
  if (tid < 128) bes[tid] = bem[tid];
  stage_bfrag(WhF, Whg, 64, 8, tid);    // [256x64]
  stage_bfrag(WemF, Wemg, 128, 2, tid); // [64x128]

  for (int idx = tid; idx < TILE * 64; idx += 256) {
    int r = idx >> 6, c4 = idx & 63;
    int gi = n0 + r;
    f32x4 v = (f32x4){0.f, 0.f, 0.f, 0.f};
    if (gi < N_) {
      if (c4 < 32) v = ((const f32x4*)x)[(size_t)gi * 32 + c4];
      else         v = ((const f32x4*)msg)[(size_t)gi * 32 + (c4 - 32)];
    }
    store_bf4(&A[r * 256 + c4 * 4], v);
  }
  __syncthreads();

  // H = A @ Wh + bh (leaky == identity)
  for (int tile = w; tile < 32; tile += 8) {
    int mt = tile >> 2, ct = tile & 3;
    v8f acc = gemm_tile(A, 256, WhF, 8, mt, ct, lane);
    int ln = lane & 15, hi = lane >> 4, n = ct * 16 + ln;
    float bv = bhs[n];
#pragma unroll
    for (int r = 0; r < 8; ++r)
      H[(mt * 16 + r + 8 * hi) * 64 + n] = (__bf16)(acc[r] + bv);
  }
  __syncthreads();

  // out = H @ W_em + b_em, stored straight to global (f32).
  for (int tile = w; tile < 64; tile += 8) {
    int mt = tile >> 3, ct = tile & 7;
    v8f acc = gemm_tile(H, 64, WemF, 2, mt, ct, lane);
    int ln = lane & 15, hi = lane >> 4, n = ct * 16 + ln;
    float bv = bes[n];
#pragma unroll
    for (int r = 0; r < 8; ++r) {
      int row = n0 + mt * 16 + r + 8 * hi;
      if (row < N_) out[(size_t)row * 128 + n] = acc[r] + bv;
    }
  }
}

// ---------------------------------------------------------------------------
extern "C" void kernel_launch(void* const* d_in, const int* in_sizes, int n_in,
                              void* d_out, int out_size, void* d_ws, size_t ws_size,
                              hipStream_t stream) {
  (void)in_sizes; (void)n_in; (void)out_size; (void)ws_size;
  const float* x     = (const float*)d_in[0];
  const float* e     = (const float*)d_in[1];
  const int*   src   = (const int*)d_in[2];
  const int*   dst   = (const int*)d_in[3];
  const float* W_nl1 = (const float*)d_in[4];
  const float* b_nl1 = (const float*)d_in[5];
  const float* W_el  = (const float*)d_in[6];
  const float* b_el  = (const float*)d_in[7];
  const float* W1    = (const float*)d_in[8];
  const float* b1    = (const float*)d_in[9];
  const float* W2    = (const float*)d_in[10];
  const float* b2    = (const float*)d_in[11];
  const float* W_nl2 = (const float*)d_in[12];
  const float* b_nl2 = (const float*)d_in[13];
  const float* W_ml  = (const float*)d_in[14];
  const float* b_ml  = (const float*)d_in[15];
  const float* W_em  = (const float*)d_in[16];
  const float* b_em  = (const float*)d_in[17];
  float* out = (float*)d_out;

  const int N_ = NNODES, E_ = NEDGES;

  // Workspace layout: msg accumulator + bf16 weights + folded biases.
  float*  msg  = (float*)d_ws;                       // N*128 f32
  __bf16* Wz   = (__bf16*)(msg + (size_t)N_ * 128);  // 256*64
  __bf16* W1b  = Wz + 256 * 64;                      // 64*64
  __bf16* W2b  = W1b + 64 * 64;                      // 64*128
  __bf16* Whb  = W2b + 64 * 128;                     // 256*64
  __bf16* Wemb = Whb + 256 * 64;                     // 64*128
  float*  bz   = (float*)(Wemb + 64 * 128);          // 64
  float*  bh   = bz + 64;                            // 64

  int n4 = N_ * 128 / 4;
  zero_msg_kernel<<<(n4 + 255) / 256, 256, 0, stream>>>((f32x4*)msg, n4);
  prep_weights_kernel<<<64, 256, 0, stream>>>(
      W_nl1, W_el, W1, W2, W_nl2, W_ml, W_em,
      b_nl1, b_el, b_nl2, b_ml,
      Wz, W1b, W2b, Whb, Wemb, bz, bh);

  edge_kernel<<<(E_ + TILE - 1) / TILE, 256, 157696, stream>>>(
      x, e, src, dst, Wz, W1b, W2b, bz, b1, b2, msg, E_);

  node_kernel<<<(N_ + TILE - 1) / TILE, 256, 131840, stream>>>(
      x, msg, Whb, Wemb, bh, b_em, out, N_);
}
